// FusedMoE_32495722561854
// MI455X (gfx1250) — matrix operand order repaired and checked by reading, not compiled, not run on hardware
//
#include <hip/hip_runtime.h>
#include <hip/hip_bf16.h>

// Problem constants (match reference)
#define T_TOK 512
#define NEXP  8
#define HDIM  2048
#define IDIM  5632
#define NROWS (2 * T_TOK)   // total (token, expert-slot) rows = 1024

typedef __attribute__((ext_vector_type(16))) __bf16 v16bf;
typedef __attribute__((ext_vector_type(8)))  __bf16 v8bf;
typedef __attribute__((ext_vector_type(8)))  float  v8f;
typedef __attribute__((ext_vector_type(4)))  float  v4f;

// ---------------------------------------------------------------------------
// 1) Routing: softmax + top-2 + renormalize.  Softmax normalization cancels in
//    the renorm, so w0 = e0/(e0+e1) with e_i = exp(l_i - max).
// ---------------------------------------------------------------------------
__global__ __launch_bounds__(256) void moe_route(const float* __restrict__ logits,
                                                 int* __restrict__ topk_id,
                                                 float* __restrict__ topk_w) {
    int t = blockIdx.x * blockDim.x + threadIdx.x;
    if (t >= T_TOK) return;
    float l[NEXP];
#pragma unroll
    for (int i = 0; i < NEXP; ++i) l[i] = logits[t * NEXP + i];
    int i0 = 0;
#pragma unroll
    for (int i = 1; i < NEXP; ++i) if (l[i] > l[i0]) i0 = i;
    int i1 = (i0 == 0) ? 1 : 0;
#pragma unroll
    for (int i = 0; i < NEXP; ++i) if (i != i0 && l[i] > l[i1]) i1 = i;
    float m  = l[i0];
    float e0 = __expf(l[i0] - m);
    float e1 = __expf(l[i1] - m);
    float inv = __builtin_amdgcn_rcpf(e0 + e1);
    topk_id[t * 2 + 0] = i0;
    topk_id[t * 2 + 1] = i1;
    topk_w[t * 2 + 0]  = e0 * inv;
    topk_w[t * 2 + 1]  = e1 * inv;
}

// ---------------------------------------------------------------------------
// 2) Histogram + exclusive scan + compacted per-expert row lists (one block).
// ---------------------------------------------------------------------------
__global__ __launch_bounds__(256) void moe_scan_fill(const int* __restrict__ topk_id,
                                                     int* __restrict__ counts,
                                                     int* __restrict__ offs,
                                                     int* __restrict__ row_tok,
                                                     int* __restrict__ slot2row) {
    __shared__ int sc[NEXP], so[NEXP], cur[NEXP];
    if (threadIdx.x < NEXP) { sc[threadIdx.x] = 0; cur[threadIdx.x] = 0; }
    __syncthreads();
    for (int s = threadIdx.x; s < NROWS; s += 256) atomicAdd(&sc[topk_id[s]], 1);
    __syncthreads();
    if (threadIdx.x == 0) {
        int o = 0;
        for (int e = 0; e < NEXP; ++e) { so[e] = o; o += sc[e]; }
    }
    __syncthreads();
    if (threadIdx.x < NEXP) {
        counts[threadIdx.x] = sc[threadIdx.x];
        offs[threadIdx.x]   = so[threadIdx.x];
    }
    for (int s = threadIdx.x; s < NROWS; s += 256) {
        int e   = topk_id[s];
        int pos = atomicAdd(&cur[e], 1);
        int row = so[e] + pos;
        row_tok[row] = s >> 1;
        slot2row[s]  = row;
    }
}

// ---------------------------------------------------------------------------
// 3) Gather selected token rows and convert fp32 -> bf16 once.
// ---------------------------------------------------------------------------
__global__ __launch_bounds__(256) void moe_pack(const float* __restrict__ x,
                                                const int* __restrict__ row_tok,
                                                __bf16* __restrict__ xpack) {
    int row = blockIdx.x;
    int t   = row_tok[row];
    const float* src = x + (size_t)t * HDIM;
    __bf16*      dst = xpack + (size_t)row * HDIM;
    for (int i = threadIdx.x; i < HDIM; i += 256) dst[i] = (__bf16)src[i];
}

// Helper: non-temporal load of 16 consecutive fp32 (streamed weights; NT keeps
// the 1.1 GB weight stream from evicting xpack/act out of the 192 MB L2),
// converted to a bf16 WMMA B-fragment.
__device__ inline v16bf load_cvt_b16_nt(const float* p) {
    v4f b0 = __builtin_nontemporal_load((const v4f*)(p + 0));
    v4f b1 = __builtin_nontemporal_load((const v4f*)(p + 4));
    v4f b2 = __builtin_nontemporal_load((const v4f*)(p + 8));
    v4f b3 = __builtin_nontemporal_load((const v4f*)(p + 12));
    v16bf f;
#pragma unroll
    for (int j = 0; j < 4; ++j) {
        f[j + 0]  = (__bf16)b0[j];
        f[j + 4]  = (__bf16)b1[j];
        f[j + 8]  = (__bf16)b2[j];
        f[j + 12] = (__bf16)b3[j];
    }
    return f;
}

// Fast SiLU via hardware exp + rcp (avoids IEEE division expansion).
__device__ inline float silu_fast(float g) {
    return g * __builtin_amdgcn_rcpf(1.0f + __expf(-g));
}

// ---------------------------------------------------------------------------
// 4) Grouped GEMM1 (gate+up) + fused SwiGLU.  Block: M=128 rows (8 waves x 16),
//    N=16 columns of I.  K = H in steps of 32 with v_wmma_f32_16x16x32_bf16.
//    A-frag: lane<16 -> row=lane, K {k..k+15}; lane>=16 -> K {k+16..k+31};
//    B-frag: lane<16 -> col=lane, K {k..k+15}; lane>=16 -> K {k+16..k+31}.
// ---------------------------------------------------------------------------
__global__ __launch_bounds__(256) void moe_gemm1(const float* __restrict__ w13,
                                                 const __bf16* __restrict__ xpack,
                                                 const int* __restrict__ counts,
                                                 const int* __restrict__ offs,
                                                 __bf16* __restrict__ act) {
    const int e    = blockIdx.y >> 2;
    const int mblk = blockIdx.y & 3;
    const int cnt  = counts[e];
    const int wave = threadIdx.x >> 5;
    const int lane = threadIdx.x & 31;
    const int p0   = mblk * 128 + wave * 16;   // expert-local first row of wave
    if (p0 >= cnt) return;                      // wave-uniform: EXEC stays full
    const int base = offs[e];
    const int n0   = blockIdx.x * 16;

    const int half = lane >> 4;
    const int r    = lane & 15;

    int arow = base + p0 + r;
    if (arow > NROWS - 1) arow = NROWS - 1;     // clamp reads; stores are masked
    const __bf16* aptr = xpack + (size_t)arow * HDIM + half * 16;

    const float* bg = w13 + ((size_t)e * (2 * IDIM) + (size_t)(n0 + r)) * HDIM + half * 16;
    const float* bu = bg + (size_t)IDIM * HDIM;

    v8f accG = {};
    v8f accU = {};
    for (int k = 0; k < HDIM; k += 32) {
        v8bf a0 = *(const v8bf*)(aptr + k);
        v8bf a1 = *(const v8bf*)(aptr + k + 8);
        v16bf afrag = __builtin_shufflevector(a0, a1, 0, 1, 2, 3, 4, 5, 6, 7,
                                              8, 9, 10, 11, 12, 13, 14, 15);
        v16bf bgf = load_cvt_b16_nt(bg + k);
        v16bf buf = load_cvt_b16_nt(bu + k);
        accG = __builtin_amdgcn_wmma_f32_16x16x32_bf16(false, afrag, false, bgf,
                                                       (short)0, accG, false, false);
        accU = __builtin_amdgcn_wmma_f32_16x16x32_bf16(false, afrag, false, buf,
                                                       (short)0, accU, false, false);
    }

    // SwiGLU and scatter-store act (bf16).  C layout: elem j -> M=j+8*half, N=r.
    __bf16* ap = act + (size_t)(base + p0 + half * 8) * IDIM + n0 + r;
    if (p0 + 16 <= cnt) {
        // Fast path: whole 16-row tile valid, no per-element masking.
#pragma unroll
        for (int j = 0; j < 8; ++j) {
            ap[(size_t)j * IDIM] = (__bf16)(silu_fast(accG[j]) * accU[j]);
        }
    } else {
#pragma unroll
        for (int j = 0; j < 8; ++j) {
            if (p0 + j + half * 8 < cnt) {
                ap[(size_t)j * IDIM] = (__bf16)(silu_fast(accG[j]) * accU[j]);
            }
        }
    }
}

// ---------------------------------------------------------------------------
// 5) Grouped GEMM2 (down projection): y[row, h] = act[row, :] . w2[e, h, :]
// ---------------------------------------------------------------------------
__global__ __launch_bounds__(256) void moe_gemm2(const float* __restrict__ w2,
                                                 const __bf16* __restrict__ act,
                                                 const int* __restrict__ counts,
                                                 const int* __restrict__ offs,
                                                 float* __restrict__ y) {
    const int e    = blockIdx.y >> 2;
    const int mblk = blockIdx.y & 3;
    const int cnt  = counts[e];
    const int wave = threadIdx.x >> 5;
    const int lane = threadIdx.x & 31;
    const int p0   = mblk * 128 + wave * 16;
    if (p0 >= cnt) return;
    const int base = offs[e];
    const int n0   = blockIdx.x * 16;           // H tile

    const int half = lane >> 4;
    const int r    = lane & 15;

    int arow = base + p0 + r;
    if (arow > NROWS - 1) arow = NROWS - 1;
    const __bf16* aptr = act + (size_t)arow * IDIM + half * 16;
    const float*  bptr = w2 + ((size_t)e * HDIM + (size_t)(n0 + r)) * IDIM + half * 16;

    v8f acc = {};
    for (int k = 0; k < IDIM; k += 32) {
        v8bf a0 = *(const v8bf*)(aptr + k);
        v8bf a1 = *(const v8bf*)(aptr + k + 8);
        v16bf afrag = __builtin_shufflevector(a0, a1, 0, 1, 2, 3, 4, 5, 6, 7,
                                              8, 9, 10, 11, 12, 13, 14, 15);
        v16bf bfrag = load_cvt_b16_nt(bptr + k);
        acc = __builtin_amdgcn_wmma_f32_16x16x32_bf16(false, afrag, false, bfrag,
                                                      (short)0, acc, false, false);
    }

    float* yp = y + (size_t)(base + p0 + half * 8) * HDIM + n0 + r;
    if (p0 + 16 <= cnt) {
#pragma unroll
        for (int j = 0; j < 8; ++j) yp[(size_t)j * HDIM] = acc[j];
    } else {
#pragma unroll
        for (int j = 0; j < 8; ++j) {
            if (p0 + j + half * 8 < cnt) yp[(size_t)j * HDIM] = acc[j];
        }
    }
}

// ---------------------------------------------------------------------------
// 6) Weighted combine: out[t] = w0*y[row(t,0)] + w1*y[row(t,1)].  Deterministic.
// ---------------------------------------------------------------------------
__global__ __launch_bounds__(256) void moe_combine(const float* __restrict__ y,
                                                   const int* __restrict__ slot2row,
                                                   const float* __restrict__ topk_w,
                                                   float* __restrict__ out) {
    int idx = blockIdx.x * blockDim.x + threadIdx.x;
    if (idx >= T_TOK * HDIM) return;
    int t = idx / HDIM;
    int h = idx - t * HDIM;
    int r0 = slot2row[t * 2 + 0];
    int r1 = slot2row[t * 2 + 1];
    out[idx] = topk_w[t * 2 + 0] * y[(size_t)r0 * HDIM + h] +
               topk_w[t * 2 + 1] * y[(size_t)r1 * HDIM + h];
}

// ---------------------------------------------------------------------------
// Launcher.  Workspace layout (bytes):
//   [0)        topk_id   int[1024]
//   [4096)     topk_w    float[1024]
//   [8192)     counts    int[8]
//   [8256)     offs      int[8]
//   [12288)    row_tok   int[1024]
//   [16384)    slot2row  int[1024]
//   [65536)    xpack     bf16[1024*2048]   (4 MB)
//   [4259840)  act       bf16[1024*5632]   (11.5 MB)
//   [15794176) y         f32 [1024*2048]   (8 MB)      total ~24.2 MB
// ---------------------------------------------------------------------------
extern "C" void kernel_launch(void* const* d_in, const int* in_sizes, int n_in,
                              void* d_out, int out_size, void* d_ws, size_t ws_size,
                              hipStream_t stream) {
    const float* x   = (const float*)d_in[0];
    const float* rl  = (const float*)d_in[1];
    const float* w13 = (const float*)d_in[2];
    const float* w2  = (const float*)d_in[3];
    // d_in[4] = top_k (== 2, hardcoded in kernels)
    float* out = (float*)d_out;

    char* ws = (char*)d_ws;
    int*    topk_id  = (int*)(ws + 0);
    float*  topk_w   = (float*)(ws + 4096);
    int*    counts   = (int*)(ws + 8192);
    int*    offs     = (int*)(ws + 8256);
    int*    row_tok  = (int*)(ws + 12288);
    int*    slot2row = (int*)(ws + 16384);
    __bf16* xpack    = (__bf16*)(ws + 65536);
    __bf16* act      = (__bf16*)(ws + 4259840);
    float*  yy       = (float*)(ws + 15794176);

    moe_route<<<(T_TOK + 255) / 256, 256, 0, stream>>>(rl, topk_id, topk_w);
    moe_scan_fill<<<1, 256, 0, stream>>>(topk_id, counts, offs, row_tok, slot2row);
    moe_pack<<<NROWS, 256, 0, stream>>>(x, row_tok, xpack);

    dim3 g1(IDIM / 16, NEXP * 4);   // 352 x 32
    moe_gemm1<<<g1, 256, 0, stream>>>(w13, xpack, counts, offs, act);

    dim3 g2(HDIM / 16, NEXP * 4);   // 128 x 32
    moe_gemm2<<<g2, 256, 0, stream>>>(w2, act, counts, offs, yy);

    moe_combine<<<(T_TOK * HDIM) / 256, 256, 0, stream>>>(yy, slot2row, topk_w, out);
}
